// TimeWiseSelfAttention_5162550690343
// MI455X (gfx1250) — matrix operand order repaired and checked by reading, not compiled, not run on hardware
//
#include <hip/hip_runtime.h>
#include <hip/hip_bf16.h>

// MI455X / gfx1250, wave32. Matrix math via V_WMMA_F32_16X16X32_BF16;
// A-tile staging via GLOBAL_LOAD_ASYNC_TO_LDS_B128 when available.

typedef __attribute__((ext_vector_type(8)))  float  f32x8;
typedef __attribute__((ext_vector_type(16))) __bf16 bf16x16;

union BFrag { bf16x16 v; unsigned int u[8]; };

__device__ __forceinline__ unsigned short f32_to_bf16(float f) {
  union { float f; unsigned int u; } c; c.f = f;
  unsigned int u = c.u;
  unsigned int r = (u + 0x7FFFu + ((u >> 16) & 1u)) >> 16;  // RNE
  return (unsigned short)r;
}

#if defined(__has_builtin)
#  if __has_builtin(__builtin_amdgcn_global_load_async_to_lds_b128)
#    define HAVE_ASYNC_COPY 1
#  endif
#endif

// clang's builtin signature uses gcc-style generic vectors: v4i addrspace(1)*/(3)*
typedef int v4i __attribute__((__vector_size__(16)));
typedef __attribute__((address_space(1))) v4i GV4;
typedef __attribute__((address_space(3))) v4i LV4;

__device__ __forceinline__ void async_copy_b128(const unsigned short* g, unsigned short* l) {
#if defined(HAVE_ASYNC_COPY)
  __builtin_amdgcn_global_load_async_to_lds_b128((GV4*)g, (LV4*)l, 0, 0);
#else
  *(uint4*)l = *(const uint4*)g;
#endif
}

__device__ __forceinline__ void async_fence() {
#if defined(HAVE_ASYNC_COPY)
#  if __has_builtin(__builtin_amdgcn_s_wait_asynccnt)
  __builtin_amdgcn_s_wait_asynccnt(0);
#  else
  asm volatile("s_wait_asynccnt 0x0" ::: "memory");
#  endif
#endif
}

#define BM 128
#define BN 64
#define BK 32
#define SP 36   // padded LDS row stride (u16): 72B rows -> 8B aligned, low-conflict

// ---------------------------------------------------------------------------
// Generic bf16 WMMA GEMM, double-buffered LDS: C = A*B (+bias)(+epilogue)
// 256 threads = 8 waves; block tile 128x64; wave: 16 rows x 64 cols.
// aHead/bHead: per-blockIdx.x element offsets (per-head batched GEMM);
// when bHead != 0, B's column window starts at 0 (per-head 64-wide B).
// EPI: 0 f32(+bias) | 1 f32, mask-> -inf | 2 bf16, mask->0 | 3 f32 +residual
// ---------------------------------------------------------------------------
template<int EPI>
__global__ __launch_bounds__(256)
void gemm_wmma(const unsigned short* __restrict__ A, int lda, long aBatch, int aHead,
               const unsigned short* __restrict__ Bm, int ldb, long bBatch, int bHead,
               void* __restrict__ Cout, int ldc, long cBatch,
               const float* __restrict__ bias,
               const int* __restrict__ mask,
               const float* __restrict__ resid,
               int K)
{
  __shared__ unsigned short As[2][BM][SP];
  __shared__ unsigned short BsT[2][BN][SP];   // B transposed: BsT[buf][n][k]
  const int tid  = threadIdx.x;
  const int wave = tid >> 5;
  const int lane = tid & 31;
  const int bz   = blockIdx.z;
  const int m0   = blockIdx.y * BM;
  const int n0   = blockIdx.x * BN;
  const unsigned short* Ab = A  + (long)bz * aBatch + (long)blockIdx.x * aHead;
  const unsigned short* Bb = Bm + (long)bz * bBatch + (long)blockIdx.x * bHead;
  const int nB0 = bHead ? 0 : n0;

  // A staging: each thread copies 16 u16 (2 x b128) of one half-row
  const int arowL = tid >> 1, ah = (tid & 1) * 16;
  const unsigned short* aSrc = Ab + (long)(m0 + arowL) * lda + ah;
  unsigned short* aDst0 = &As[0][arowL][ah];
  unsigned short* aDst1 = &As[1][arowL][ah];
  // B staging: each thread reads 8 u16 of row kk, writes transposed
  const int kkL = tid >> 3, ngL = (tid & 7) * 8;
  const unsigned short* bSrc = Bb + (long)kkL * ldb + nB0 + ngL;

  f32x8 acc[4] = {};
  const int half = lane >> 4;
  const int arow = wave * 16 + (lane & 15);
  const int nK = K / BK;

  // stage tile 0
  async_copy_b128(aSrc, aDst0);
  async_copy_b128(aSrc + 8, aDst0 + 8);
  {
    unsigned short tmp[8];
    *(uint2*)&tmp[0] = *(const uint2*)(bSrc);
    *(uint2*)&tmp[4] = *(const uint2*)(bSrc + 4);
    #pragma unroll
    for (int i = 0; i < 8; ++i) BsT[0][ngL + i][kkL] = tmp[i];
  }
  async_fence();
  __syncthreads();

  for (int kt = 0; kt < nK; ++kt) {
    const int cur = kt & 1;
    if (kt + 2 < nK) {   // prefetch two tiles ahead into L2/WGP$
      __builtin_prefetch(aSrc + (kt + 2) * BK, 0, 1);
      __builtin_prefetch(bSrc + (long)(kt + 2) * BK * ldb, 0, 1);
    }
    if (kt + 1 < nK) {   // stage next tile into the other buffer
      const int nxt = cur ^ 1;
      const unsigned short* as = aSrc + (kt + 1) * BK;
      unsigned short* ad = cur ? aDst0 : aDst1;
      async_copy_b128(as, ad);
      async_copy_b128(as + 8, ad + 8);
      const unsigned short* bs = bSrc + (long)(kt + 1) * BK * ldb;
      unsigned short tmp[8];
      *(uint2*)&tmp[0] = *(const uint2*)(bs);
      *(uint2*)&tmp[4] = *(const uint2*)(bs + 4);
      #pragma unroll
      for (int i = 0; i < 8; ++i) BsT[nxt][ngL + i][kkL] = tmp[i];
    }
    // compute current tile
    BFrag a;
    #pragma unroll
    for (int i = 0; i < 8; ++i) {
      int kb = (i >> 2) * 16 + half * 8 + (i & 3) * 2;
      a.u[i] = *(const unsigned int*)&As[cur][arow][kb];
    }
    BFrag b[4];
    #pragma unroll
    for (int j = 0; j < 4; ++j) {
      int brow = j * 16 + (lane & 15);
      #pragma unroll
      for (int i = 0; i < 8; ++i)
        b[j].u[i] = *(const unsigned int*)&BsT[cur][brow][(i >> 2) * 16 + half * 8 + (i & 3) * 2];
    }
    #pragma unroll
    for (int j = 0; j < 4; ++j)
      acc[j] = __builtin_amdgcn_wmma_f32_16x16x32_bf16(false, a.v, false, b[j].v,
                                                       (short)0, acc[j], false, false);
    if (kt + 1 < nK) async_fence();
    __syncthreads();
  }

  float*          Cf = (float*)Cout          + (long)bz * cBatch;
  unsigned short* Ch = (unsigned short*)Cout + (long)bz * cBatch;
  const int rbase = m0 + wave * 16 + (lane >> 4) * 8;
  const int cbase = n0 + (lane & 15);
  #pragma unroll
  for (int j = 0; j < 4; ++j) {
    int col = cbase + j * 16;
    float bv = bias ? bias[col] : 0.0f;
    #pragma unroll
    for (int r = 0; r < 8; ++r) {
      long row = rbase + r;
      float val = acc[j][r] + bv;
      if (EPI == 0) {
        Cf[row * ldc + col] = val;
      } else if (EPI == 1) {
        Cf[row * ldc + col] = mask[row] ? -__builtin_inff() : val;
      } else if (EPI == 2) {
        Ch[row * ldc + col] = f32_to_bf16(mask[row] ? 0.0f : val);
      } else {
        Cf[row * ldc + col] = val + resid[row * ldc + col];
      }
    }
  }
}

// ---------------------------------------------------------------------------
// attn[e,d] = sum_s k_sm[s,e] * v[s,d] per (b,h), split 8-ways over s for
// parallelism; partial results accumulated with global_atomic_add_f32 into
// attF[bh][e][d] (pre-zeroed).
// ---------------------------------------------------------------------------
#define ATT_SPLIT 8
__global__ __launch_bounds__(128)
void attn_kernel(const unsigned short* __restrict__ Ksm,
                 const unsigned short* __restrict__ Vb,
                 float* __restrict__ attF,
                 int T, int D, int H)
{
  __shared__ unsigned short kT[64][SP];  // kT[e][s]
  __shared__ unsigned short vT[64][SP];  // vT[d][s]
  const int bh = blockIdx.x;
  const int b = bh / H, h = bh % H;
  const int tid = threadIdx.x, wave = tid >> 5, lane = tid & 31;
  const int half = lane >> 4;
  const long base = (long)b * T * D + h * 64;
  const int sBeg = blockIdx.y * (T / ATT_SPLIT);
  const int sEnd = sBeg + (T / ATT_SPLIT);
  f32x8 acc[4] = {};
  for (int s0 = sBeg; s0 < sEnd; s0 += 32) {
    int s = tid >> 2, eg = tid & 3;
    const uint2* ks = (const uint2*)(Ksm + base + (long)(s0 + s) * D + eg * 16);
    const uint2* vs = (const uint2*)(Vb  + base + (long)(s0 + s) * D + eg * 16);
    if (s0 + 32 < sEnd) {
      __builtin_prefetch(Ksm + base + (long)(s0 + 32 + s) * D + eg * 16, 0, 1);
      __builtin_prefetch(Vb  + base + (long)(s0 + 32 + s) * D + eg * 16, 0, 1);
    }
    unsigned short kv[16], vv[16];
    #pragma unroll
    for (int j = 0; j < 4; ++j) { *(uint2*)&kv[j * 4] = ks[j]; *(uint2*)&vv[j * 4] = vs[j]; }
    #pragma unroll
    for (int i = 0; i < 16; ++i) { kT[eg * 16 + i][s] = kv[i]; vT[eg * 16 + i][s] = vv[i]; }
    __syncthreads();
    BFrag a;
    int arow = wave * 16 + (lane & 15);
    #pragma unroll
    for (int i = 0; i < 8; ++i) {
      int kb = (i >> 2) * 16 + half * 8 + (i & 3) * 2;
      a.u[i] = *(const unsigned int*)&kT[arow][kb];
    }
    #pragma unroll
    for (int j = 0; j < 4; ++j) {
      BFrag bb;
      int brow = j * 16 + (lane & 15);
      #pragma unroll
      for (int i = 0; i < 8; ++i)
        bb.u[i] = *(const unsigned int*)&vT[brow][(i >> 2) * 16 + half * 8 + (i & 3) * 2];
      acc[j] = __builtin_amdgcn_wmma_f32_16x16x32_bf16(false, a.v, false, bb.v,
                                                       (short)0, acc[j], false, false);
    }
    __syncthreads();
  }
  const int ebase = wave * 16 + (lane >> 4) * 8;
  #pragma unroll
  for (int j = 0; j < 4; ++j) {
    int d = j * 16 + (lane & 15);
    #pragma unroll
    for (int r = 0; r < 8; ++r)
      atomicAdd(&attF[((long)bh * 64 + (ebase + r)) * 64 + d], acc[j][r]);
  }
}

__global__ __launch_bounds__(256)
void fill_f32_kernel(float* __restrict__ p, long n)
{
  long i = (long)blockIdx.x * 256 + threadIdx.x;
  if (i < n) p[i] = 0.0f;
}

// attnT[bh][d][e] = bf16(attF[bh][e][d])
__global__ __launch_bounds__(256)
void att_cvt_kernel(const float* __restrict__ attF, unsigned short* __restrict__ attnT)
{
  int idx = blockIdx.x * 256 + threadIdx.x;   // over B*H*64*64
  int bh = idx >> 12;
  int d  = (idx >> 6) & 63;
  int e  = idx & 63;
  attnT[idx] = f32_to_bf16(attF[(bh << 12) + (e << 6) + d]);
}

// softmax over the 64-feature head dim; one wave per (row, head) -> bf16
__global__ __launch_bounds__(256)
void qsoftmax_kernel(const float* __restrict__ Q, unsigned short* __restrict__ Qs,
                     int D, int H)
{
  long w = (long)blockIdx.x * 8 + (threadIdx.x >> 5);
  int lane = threadIdx.x & 31;
  long row = w / H; int h = (int)(w % H);
  const float* p = Q + row * D + h * 64;
  float a = p[lane], b = p[lane + 32];
  float m = fmaxf(a, b);
  #pragma unroll
  for (int off = 16; off; off >>= 1) m = fmaxf(m, __shfl_xor(m, off, 32));
  float ea = __expf(a - m), eb = __expf(b - m);
  float s = ea + eb;
  #pragma unroll
  for (int off = 16; off; off >>= 1) s += __shfl_xor(s, off, 32);
  float inv = 1.0f / s;
  unsigned short* o = Qs + row * D + h * 64;
  o[lane]      = f32_to_bf16(ea * inv);
  o[lane + 32] = f32_to_bf16(eb * inv);
}

// softmax over the time dim per (b, col); masked entries are -inf -> exp 0
__global__ __launch_bounds__(256)
void ksoftmax_kernel(const float* __restrict__ Kf, unsigned short* __restrict__ Ks,
                     int T, int D)
{
  __shared__ float red[4][64];
  const int b = blockIdx.y;
  const int c = threadIdx.x & 63;
  const int part = threadIdx.x >> 6;
  const int col = blockIdx.x * 64 + c;
  const float* src = Kf + (long)b * T * D + col;
  float m = -__builtin_inff();
  for (int t = part; t < T; t += 4) m = fmaxf(m, src[(long)t * D]);
  red[part][c] = m;
  __syncthreads();
  m = fmaxf(fmaxf(red[0][c], red[1][c]), fmaxf(red[2][c], red[3][c]));
  float mm = (m > -1e37f) ? m : 0.0f;   // all-masked column -> all zeros
  __syncthreads();
  float s = 0.0f;
  for (int t = part; t < T; t += 4) s += __expf(src[(long)t * D] - mm);
  red[part][c] = s;
  __syncthreads();
  s = red[0][c] + red[1][c] + red[2][c] + red[3][c];
  float inv = (s > 0.0f) ? 1.0f / s : 0.0f;
  unsigned short* dst = Ks + (long)b * T * D + col;
  for (int t = part; t < T; t += 4)
    dst[(long)t * D] = f32_to_bf16(__expf(src[(long)t * D] - mm) * inv);
}

// MODE 0: bf16(LN(x)*gamma+beta).  MODE 1: FiLM: silu(LN(y)*(1+scale)+shift)
template<int MODE>
__global__ __launch_bounds__(256)
void layernorm_kernel(const float* __restrict__ X,
                      const float* __restrict__ gamma, const float* __restrict__ beta,
                      const float* __restrict__ filmEO,
                      unsigned short* __restrict__ Out, int D, int T)
{
  const long row = blockIdx.x;
  const int tid = threadIdx.x;
  const float* x = X + row * D;
  float v[4]; float s = 0.f, ss = 0.f;
  #pragma unroll
  for (int i = 0; i < 4; ++i) { v[i] = x[tid + i * 256]; s += v[i]; ss += v[i] * v[i]; }
  #pragma unroll
  for (int off = 16; off; off >>= 1) { s += __shfl_xor(s, off, 32); ss += __shfl_xor(ss, off, 32); }
  __shared__ float rs[8], rss[8];
  if ((tid & 31) == 0) { rs[tid >> 5] = s; rss[tid >> 5] = ss; }
  __syncthreads();
  s = 0.f; ss = 0.f;
  #pragma unroll
  for (int wv = 0; wv < 8; ++wv) { s += rs[wv]; ss += rss[wv]; }
  const float invD = 1.0f / (float)D;
  float mu = s * invD;
  float var = ss * invD - mu * mu;
  float rstd = rsqrtf(var + 1e-5f);
  unsigned short* o = Out + row * D;
  const int bidx = (int)(row / T);
  #pragma unroll
  for (int i = 0; i < 4; ++i) {
    int c = tid + i * 256;
    float ln = (v[i] - mu) * rstd * gamma[c] + beta[c];
    if (MODE == 0) {
      o[c] = f32_to_bf16(ln);
    } else {
      float g  = 1.0f + filmEO[(long)bidx * 2 * D + c];
      float sh = filmEO[(long)bidx * 2 * D + D + c];
      float hh = ln * g + sh;
      o[c] = f32_to_bf16(hh / (1.0f + __expf(-hh)));   // silu
    }
  }
}

// emb_out = silu(emb) @ film_emb_W + b   (tiny: B x 2D)
__global__ __launch_bounds__(256)
void film_emb_kernel(const float* __restrict__ emb, const float* __restrict__ W,
                     const float* __restrict__ bias, float* __restrict__ out,
                     int E, int twoD)
{
  __shared__ float se[2048];
  const int b = blockIdx.y;
  const int j = blockIdx.x * 256 + threadIdx.x;
  for (int e = threadIdx.x; e < E; e += 256) {
    float vv = emb[(long)b * E + e];
    se[e] = vv / (1.0f + __expf(-vv));
  }
  __syncthreads();
  float acc = bias[j];
  for (int e = 0; e < E; ++e) acc = fmaf(se[e], W[(long)e * twoD + j], acc);
  out[(long)b * twoD + j] = acc;
}

__global__ __launch_bounds__(256)
void cvt_bf16_kernel(const float* __restrict__ in, unsigned short* __restrict__ out, long n)
{
  long i = (long)blockIdx.x * 256 + threadIdx.x;
  if (i < n) out[i] = f32_to_bf16(in[i]);
}

extern "C" void kernel_launch(void* const* d_in, const int* in_sizes, int n_in,
                              void* d_out, int out_size, void* d_ws, size_t ws_size,
                              hipStream_t stream)
{
  (void)in_sizes; (void)n_in; (void)out_size; (void)ws_size;
  const int B = 4, T = 4096, D = 1024, E = 2048, H = 16;
  const long M = (long)B * T;

  const float* x   = (const float*)d_in[0];
  const float* emb = (const float*)d_in[1];
  const int*   msk = (const int*)d_in[2];
  const float* nsc = (const float*)d_in[3];
  const float* nbi = (const float*)d_in[4];
  const float* Wq  = (const float*)d_in[5];
  const float* bq  = (const float*)d_in[6];
  const float* Wk  = (const float*)d_in[7];
  const float* bk  = (const float*)d_in[8];
  const float* Wv  = (const float*)d_in[9];
  const float* bv  = (const float*)d_in[10];
  const float* feW = (const float*)d_in[11];
  const float* feb = (const float*)d_in[12];
  const float* fns = (const float*)d_in[13];
  const float* fnb = (const float*)d_in[14];
  const float* foW = (const float*)d_in[15];
  const float* fob = (const float*)d_in[16];

  // Workspace layout (~267 MB peak; buffers reused once dead):
  char* ws = (char*)d_ws;
  unsigned short* XN  = (unsigned short*)(ws + 0);             // 32 MB (xn bf16)
  unsigned short* WQB = (unsigned short*)(ws + (32ll  << 20)); // 2 MB
  unsigned short* WKB = (unsigned short*)(ws + (34ll  << 20)); // 2 MB
  unsigned short* WVB = (unsigned short*)(ws + (36ll  << 20)); // 2 MB
  unsigned short* WOB = (unsigned short*)(ws + (38ll  << 20)); // 2 MB
  float*          QF  = (float*)(ws + (40ll  << 20));          // 64 MB (q f32)
  float*          KF  = (float*)(ws + (104ll << 20));          // 64 MB (k f32, -inf masked)
  unsigned short* VB  = (unsigned short*)(ws + (168ll << 20)); // 32 MB (v bf16, 0 masked)
  unsigned short* QS  = (unsigned short*)(ws + (200ll << 20)); // 32 MB (softmax(q) bf16)
  unsigned short* KS  = (unsigned short*)(ws + (232ll << 20)); // 32 MB (softmax(k) bf16)
  unsigned short* ATT = (unsigned short*)(ws + (264ll << 20)); // 512 KB (attn^T bf16)
  float*          EO  = (float*)(ws + (265ll << 20));          // 32 KB (FiLM scale/shift)
  float*          ATF = (float*)(ws + (266ll << 20));          // 1 MB (attn f32 partials)
  float*          YF  = QF;   // reuse: q f32 dead after qsoftmax
  unsigned short* SIL = XN;   // reuse: xn dead after QKV GEMMs

  // 1) fp32 -> bf16 weight conversion
  const long nW = (long)D * D;
  cvt_bf16_kernel<<<dim3((unsigned)((nW + 255) / 256)), 256, 0, stream>>>(Wq,  WQB, nW);
  cvt_bf16_kernel<<<dim3((unsigned)((nW + 255) / 256)), 256, 0, stream>>>(Wk,  WKB, nW);
  cvt_bf16_kernel<<<dim3((unsigned)((nW + 255) / 256)), 256, 0, stream>>>(Wv,  WVB, nW);
  cvt_bf16_kernel<<<dim3((unsigned)((nW + 255) / 256)), 256, 0, stream>>>(foW, WOB, nW);

  // 2) xn = LN(x) -> bf16
  layernorm_kernel<0><<<dim3((unsigned)M), 256, 0, stream>>>(x, nsc, nbi, nullptr, XN, D, T);

  // 3) Q/K/V GEMMs (16384x1024 @ 1024x1024, bf16 WMMA, f32 accum)
  dim3 gGemm(D / BN, (unsigned)(M / BM), 1);
  gemm_wmma<0><<<gGemm, 256, 0, stream>>>(XN, D, 0, 0, WQB, D, 0, 0, QF, D, 0, bq, nullptr, nullptr, D);
  gemm_wmma<1><<<gGemm, 256, 0, stream>>>(XN, D, 0, 0, WKB, D, 0, 0, KF, D, 0, bk, msk, nullptr, D);
  gemm_wmma<2><<<gGemm, 256, 0, stream>>>(XN, D, 0, 0, WVB, D, 0, 0, VB, D, 0, bv, msk, nullptr, D);

  // 4) softmaxes
  qsoftmax_kernel<<<dim3((unsigned)(M * H / 8)), 256, 0, stream>>>(QF, QS, D, H);
  ksoftmax_kernel<<<dim3(D / 64, B), 256, 0, stream>>>(KF, KS, T, D);

  // 5) attn = k_sm^T @ v per (b,h), 8-way split-K with f32 atomics
  const long nAtt = (long)B * H * 64 * 64;
  fill_f32_kernel<<<dim3((unsigned)((nAtt + 255) / 256)), 256, 0, stream>>>(ATF, nAtt);
  attn_kernel<<<dim3(B * H, ATT_SPLIT), 128, 0, stream>>>(KS, VB, ATF, T, D, H);
  att_cvt_kernel<<<dim3((unsigned)(nAtt / 256)), 256, 0, stream>>>(ATF, ATT);

  // 6) y = q_sm @ attn^T  (per b batched; head via blockIdx.x head-strides)
  dim3 gy(H, (unsigned)(T / BM), B);
  gemm_wmma<0><<<gy, 256, 0, stream>>>(QS, D, (long)T * D, 64,
                                       ATT, 64, (long)H * 64 * 64, 64 * 64,
                                       YF, D, (long)T * D, nullptr, nullptr, nullptr, 64);

  // 7) FiLM scale/shift from emb
  film_emb_kernel<<<dim3(2 * D / 256, B), 256, 0, stream>>>(emb, feW, feb, EO, E, 2 * D);

  // 8) s = silu(LN(y)*(1+scale)+shift) -> bf16
  layernorm_kernel<1><<<dim3((unsigned)M), 256, 0, stream>>>(YF, fns, fnb, EO, SIL, D, T);

  // 9) out = x + s @ film_out_W + b
  gemm_wmma<3><<<gGemm, 256, 0, stream>>>(SIL, D, 0, 0, WOB, D, 0, 0, (float*)d_out, D, 0,
                                          fob, nullptr, x, D);
}
